// MemEffTokenCreation_24592982737088
// MI455X (gfx1250) — matrix operand order repaired and checked by reading, not compiled, not run on hardware
//
#include <hip/hip_runtime.h>
#include <hip/hip_bf16.h>

// ---------------------------------------------------------------------------
// MemEffTokenCreation for MI455X (gfx1250, wave32, WMMA 16x16x32 f16).
//   1. convert weights / q*scale to f16
//   2. layernorm(x) -> xn_f16
//   3. GEMM: kv = xn @ kv_w^T              (100352x384 @ 384x768)
//   4. attn1 (196x196 per b,h, softmax)    -> q_cond0
//   5. GEMM fc1 (+bias, exact GELU), GEMM fc2 (+bias) -> q_cond
//   6. attn2 scores (head-mean in LDS) -> argmax -> centroid -> mask bitmap
//   7. flash attention over N=3136 (bitmap mask, TDM-staged K) -> attn_out
//   8. GEMM proj (+bias) -> d_out (f32)
// ---------------------------------------------------------------------------

typedef __attribute__((ext_vector_type(8)))  _Float16 v8h;
typedef __attribute__((ext_vector_type(16))) _Float16 v16h;
typedef __attribute__((ext_vector_type(8)))  float    v8f;
typedef __attribute__((ext_vector_type(4)))  unsigned int u32x4;
typedef __attribute__((ext_vector_type(8)))  int      i32x8;
typedef __attribute__((ext_vector_type(4)))  int      i32x4;

#define BATCH  32
#define NPOS   3136
#define NTOK   196
#define CDIM   384
#define HEADS  8
#define HDIM   48
#define QK_SCALE 0.14433756729740645f   // 1/sqrt(48)
#define NEG_BIG  (-1.0e9f)
#define MASK_WORDS 98                    // 3136 bits / 32

#ifdef __has_builtin
#if __has_builtin(__builtin_amdgcn_tensor_load_to_lds)
#define HAVE_TDM 1
#endif
#endif
#ifndef HAVE_TDM
#define HAVE_TDM 0
#endif

// Compile-probe diagnostics (non-fatal; confirm which staging path is built).
#if HAVE_TDM
#if __clang_major__ >= 23
#warning "attn3 K staging: TDM tensor_load_to_lds (6-arg clang>=23 form) enabled"
#else
#warning "attn3 K staging: TDM tensor_load_to_lds (5-arg ROCm7.2 form) enabled"
#endif
#else
#warning "attn3 K staging: TDM builtin NOT available -> b128 vector-load fallback"
#endif

static __device__ __forceinline__ v8f vzero8() {
  v8f z;
#pragma unroll
  for (int i = 0; i < 8; ++i) z[i] = 0.0f;
  return z;
}

static __device__ __forceinline__ v16h cat8(v8h lo, v8h hi) {
  return __builtin_shufflevector(lo, hi, 0,1,2,3,4,5,6,7,8,9,10,11,12,13,14,15);
}

// A fragment (16x32 f16), per ISA 7.12.2: lane m = lane&15, K groups split at
// lane 16; element i -> k = (i<8?0:16) + ((lane&16)?8:0) + (i&7).
static __device__ __forceinline__ v16h frag_a16(const _Float16* base, int lda, int lane) {
  const int m  = lane & 15;
  const int kb = (lane & 16) ? 8 : 0;
  const _Float16* p = base + m * lda + kb;
  v8h lo = *(const v8h*)(p);
  v8h hi = *(const v8h*)(p + 16);
  return cat8(lo, hi);
}

// B fragment (32x16 f16) from storage laid out [n_out][k]:
// lane column n = lane&15; element i -> k = (i<8?0:16) + ((lane&16)?8:0) + (i&7).
static __device__ __forceinline__ v16h frag_b16(const _Float16* base, int lda, int lane) {
  const int n  = lane & 15;
  const int kb = (lane & 16) ? 8 : 0;
  const _Float16* p = base + n * lda + kb;
  v8h lo = *(const v8h*)(p);
  v8h hi = *(const v8h*)(p + 16);
  return cat8(lo, hi);
}

static __device__ __forceinline__ v8f wmma_f16(v16h a, v16h b, v8f c) {
  return __builtin_amdgcn_wmma_f32_16x16x32_f16(false, a, false, b, (short)0, c, false, false);
}

#if HAVE_TDM
// Issue a 2-D TDM load: tile_h rows x tile_w f16 elements into LDS (contiguous,
// row stride = tile_w elements).  Elements with x >= tensor_w read as zero
// (TDM OOB rule), giving free zero padding.  D# bitfields per ISA ch.8.
static __device__ __forceinline__ void tdm_load_2d_f16(
    unsigned lds_addr, unsigned long long gaddr,
    unsigned tile_w, unsigned tile_h, unsigned tensor_w, unsigned row_stride) {
  u32x4 g0;
  g0[0] = 1u;                                               // count=1 user D#
  g0[1] = lds_addr;                                         // LDS byte address
  g0[2] = (unsigned)gaddr;                                  // global addr lo
  g0[3] = ((unsigned)(gaddr >> 32) & 0x01FFFFFFu) | (2u << 30);  // hi | type=2
  i32x8 g1;
  g1[0] = (int)(1u << 16);                 // wg_mask=0, data_size=1 (2 bytes)
  g1[1] = (int)(tensor_w << 16);           // tensor_dim0[15:0] at bits 63:48
  g1[2] = (int)((tensor_w >> 16) | (tile_h << 16));  // dim0 hi | tensor_dim1 lo
  g1[3] = (int)((tile_h >> 16) | (tile_w << 16));    // dim1 hi | tile_dim0
  g1[4] = (int)tile_h;                     // tile_dim1 (tile_dim2 = 0)
  g1[5] = (int)row_stride;                 // tensor_dim0_stride lo32
  g1[6] = 0;                               // stride hi | tensor_dim1_stride lo
  g1[7] = 0;
  i32x4 gz;
  gz[0] = 0; gz[1] = 0; gz[2] = 0; gz[3] = 0;               // unused dims
#if __clang_major__ >= 23
  i32x8 gz8;
#pragma unroll
  for (int i = 0; i < 8; ++i) gz8[i] = 0;
  __builtin_amdgcn_tensor_load_to_lds(g0, g1, gz, gz, gz8, 0);
#else
  __builtin_amdgcn_tensor_load_to_lds(g0, g1, gz, gz, 0);
#endif
}
#endif

// ---------------------------------------------------------------------------
// f32 -> f16 convert (optionally scaled)
// ---------------------------------------------------------------------------
__global__ __launch_bounds__(256)
void cvt_f32_to_f16(const float* __restrict__ src, _Float16* __restrict__ dst,
                    int n, float scale) {
  int i = blockIdx.x * 256 + threadIdx.x;
  if (i < n) dst[i] = (_Float16)(src[i] * scale);
}

// ---------------------------------------------------------------------------
// LayerNorm over C=384, one block (128 threads) per row; output f16.
// ---------------------------------------------------------------------------
__global__ __launch_bounds__(128)
void ln_kernel(const float* __restrict__ x, const float* __restrict__ w,
               const float* __restrict__ b, _Float16* __restrict__ xn) {
  const int row = blockIdx.x;
  const int tid = threadIdx.x;
  const float* px = x + (size_t)row * CDIM;
  __shared__ float red[128];

  float v3[3];
  float s = 0.0f;
#pragma unroll
  for (int i = 0; i < 3; ++i) { v3[i] = px[tid + i * 128]; s += v3[i]; }
  red[tid] = s;
  __syncthreads();
  for (int off = 64; off > 0; off >>= 1) {
    if (tid < off) red[tid] += red[tid + off];
    __syncthreads();
  }
  const float mu = red[0] * (1.0f / CDIM);
  __syncthreads();
  float vs = 0.0f;
#pragma unroll
  for (int i = 0; i < 3; ++i) { float d = v3[i] - mu; vs += d * d; }
  red[tid] = vs;
  __syncthreads();
  for (int off = 64; off > 0; off >>= 1) {
    if (tid < off) red[tid] += red[tid + off];
    __syncthreads();
  }
  const float inv = rsqrtf(red[0] * (1.0f / CDIM) + 1e-6f);
#pragma unroll
  for (int i = 0; i < 3; ++i) {
    int c = tid + i * 128;
    xn[(size_t)row * CDIM + c] = (_Float16)((v3[i] - mu) * inv * w[c] + b[c]);
  }
}

// ---------------------------------------------------------------------------
// Generic WMMA GEMM: C[M,Nout] = A[M,K](f16) * W[Nout,K]^T(f16) (+bias)(+gelu)
// Requires M%128==0, K%64==0, Nout%64==0.  Block 256 = 8 waves (4x2),
// each wave computes a 32x32 output block; K tile = 64 (8 WMMA per sync).
// ---------------------------------------------------------------------------
__global__ __launch_bounds__(256)
void gemm_nt_f16(const _Float16* __restrict__ A, const _Float16* __restrict__ W,
                 const float* __restrict__ bias, _Float16* __restrict__ Of16,
                 float* __restrict__ Of32, int M, int K, int Nout, int do_gelu) {
  __shared__ _Float16 lA[128 * 72];
  __shared__ _Float16 lB[64 * 72];
  const int tid  = threadIdx.x;
  const int lane = tid & 31;
  const int wave = tid >> 5;
  const int wm = wave >> 1, wn = wave & 1;
  const int m0 = blockIdx.y * 128;
  const int n0 = blockIdx.x * 64;

  v8f acc[2][2];
  acc[0][0] = vzero8(); acc[0][1] = vzero8();
  acc[1][0] = vzero8(); acc[1][1] = vzero8();

  for (int k0 = 0; k0 < K; k0 += 64) {
    { // A tile 128x64: 2 threads/row, 32 halves each
      int row = tid >> 1;
      int h32 = (tid & 1) * 32;
      const uint4* src = (const uint4*)(A + (size_t)(m0 + row) * K + k0 + h32);
      uint4* dst = (uint4*)(lA + row * 72 + h32);
      dst[0] = src[0]; dst[1] = src[1]; dst[2] = src[2]; dst[3] = src[3];
      if (k0 + 64 < K)
        __builtin_prefetch(A + (size_t)(m0 + row) * K + k0 + 64 + h32, 0, 1);
    }
    { // W tile 64x64: 4 threads/row, 16 halves each
      int row = tid >> 2;
      int h16 = (tid & 3) * 16;
      const uint4* src = (const uint4*)(W + (size_t)(n0 + row) * K + k0 + h16);
      uint4* dst = (uint4*)(lB + row * 72 + h16);
      dst[0] = src[0]; dst[1] = src[1];
    }
    __syncthreads();
#pragma unroll
    for (int ks = 0; ks < 2; ++ks) {
      v16h af[2], bf[2];
#pragma unroll
      for (int t = 0; t < 2; ++t)
        af[t] = frag_a16(lA + (wm * 32 + t * 16) * 72 + ks * 32, 72, lane);
#pragma unroll
      for (int t = 0; t < 2; ++t)
        bf[t] = frag_b16(lB + (wn * 32 + t * 16) * 72 + ks * 32, 72, lane);
#pragma unroll
      for (int tm = 0; tm < 2; ++tm)
#pragma unroll
        for (int tn = 0; tn < 2; ++tn)
          acc[tm][tn] = wmma_f16(af[tm], bf[tn], acc[tm][tn]);
    }
    __syncthreads();
  }

  const int mhi = (lane & 16) ? 8 : 0;
  const int nc  = lane & 15;
#pragma unroll
  for (int tm = 0; tm < 2; ++tm)
#pragma unroll
    for (int tn = 0; tn < 2; ++tn) {
#pragma unroll
      for (int r = 0; r < 8; ++r) {
        int mm = m0 + wm * 32 + tm * 16 + r + mhi;
        int nn = n0 + wn * 32 + tn * 16 + nc;
        float v = acc[tm][tn][r];
        if (bias) v += bias[nn];
        if (do_gelu) v = 0.5f * v * (1.0f + erff(v * 0.70710678118654752f));
        size_t off = (size_t)mm * Nout + nn;
        if (Of16) Of16[off] = (_Float16)v;
        else      Of32[off] = v;
      }
    }
}

// ---------------------------------------------------------------------------
// Vectorized tile loaders (16B-aligned b128 copies with zero padding)
// ---------------------------------------------------------------------------
// Q tile 16x64 (cols >= 48 zero, rows >= NTOK zero); 128 threads.
static __device__ __forceinline__ void load_q_tile(_Float16* lQ,
                                                   const _Float16* src_base,
                                                   int m0, int tid) {
  const uint4 z = {0u, 0u, 0u, 0u};
  int row = tid >> 3, c8 = (tid & 7) * 8;
  int m = m0 + row;
  uint4 v = z;
  if (m < NTOK && c8 < HDIM)
    v = *(const uint4*)(src_base + (size_t)m * CDIM + c8);
  *(uint4*)(lQ + row * 64 + c8) = v;
}

// Gathered K tile 16x64 loaded by ONE wave (lane pairs per row).
static __device__ __forceinline__ void load_k_tile_gather(
    _Float16* lKw, const _Float16* __restrict__ kvf, const int* __restrict__ idx,
    int b, int h, int n_base, int lane) {
  const uint4 z = {0u, 0u, 0u, 0u};
  int row = lane >> 1;
  int n = n_base + row;
  uint4 va = z, vb = z;
  if (n < NTOK) {
    const uint4* src =
        (const uint4*)(kvf + ((size_t)b * NPOS + idx[n]) * (2 * CDIM) + h * HDIM);
    if (lane & 1) { va = src[2]; }            // cols 16..23 -> LDS 32..39 (40..47 zero, 48..63 unwritten)
    else          { va = src[0]; vb = src[1]; }  // cols 0..15
  }
  uint4* dst = (uint4*)(lKw + row * 64 + (lane & 1) * 32);
  dst[0] = va; dst[1] = vb;
}

static __device__ __forceinline__ v8f score_tile(const _Float16* lQ,
                                                 const _Float16* lKw, int lane) {
  v8f s = vzero8();
#pragma unroll
  for (int ks = 0; ks < 2; ++ks) {
    v16h a  = frag_a16(lQ + ks * 32, 64, lane);
    v16h bb = frag_b16(lKw + ks * 32, 64, lane);
    s = wmma_f16(a, bb, s);
  }
  return s;
}

// ---------------------------------------------------------------------------
// attn1: softmax((q*scale) @ k_sub^T) @ v_sub  -> q_cond0 (f16)
// grid (13 m-tiles, H, B), block 128 (4 waves).
// ---------------------------------------------------------------------------
__global__ __launch_bounds__(128)
void attn1_kernel(const _Float16* __restrict__ qs,      // 196x384 (pre-scaled)
                  const _Float16* __restrict__ kvf,     // (B*N)x768, k|v
                  const int* __restrict__ idx_sub,
                  _Float16* __restrict__ qc0) {         // (B*196)x384
  const int m0 = blockIdx.x * 16;
  const int h  = blockIdx.y;
  const int b  = blockIdx.z;
  const int tid = threadIdx.x, lane = tid & 31, wave = tid >> 5;

  __shared__ _Float16 lQ[16 * 64];
  __shared__ _Float16 lK[4][16 * 64];
  __shared__ float    lS[16 * 224];
  __shared__ _Float16 lP[16 * 224];
  __shared__ _Float16 lVt[48 * 40];

  load_q_tile(lQ, qs + h * HDIM, m0, tid);
  __syncthreads();

  // ---- scores: 13 n-tiles in groups of 4 (one per wave) ----
  for (int g = 0; g < 4; ++g) {
    int tn = g * 4 + wave;
    if (tn < 13) {
      load_k_tile_gather(lK[wave], kvf, idx_sub, b, h, tn * 16, lane);
      v8f s = score_tile(lQ, lK[wave], lane);
      const int ncol = tn * 16 + (lane & 15);
      const int mr   = (lane & 16) ? 8 : 0;
#pragma unroll
      for (int r = 0; r < 8; ++r) lS[(r + mr) * 224 + ncol] = s[r];
    }
    __syncthreads();
  }

  // ---- softmax over n<196 ----
  if (tid < 16) {
    float mx = -1e30f;
    for (int n = 0; n < NTOK; ++n) mx = fmaxf(mx, lS[tid * 224 + n]);
    float sum = 0.0f;
    for (int n = 0; n < 224; ++n) {
      float p = (n < NTOK) ? __expf(lS[tid * 224 + n] - mx) : 0.0f;
      lS[tid * 224 + n] = p;
      sum += p;
    }
    float inv = 1.0f / sum;
    for (int n = 0; n < 224; ++n)
      lP[tid * 224 + n] = (_Float16)(lS[tid * 224 + n] * inv);
  }
  __syncthreads();

  // ---- O = P @ V_sub (K padded to 224), d-tile per wave (3 waves) ----
  v8f o = vzero8();
  for (int ks = 0; ks < 7; ++ks) {
    for (int e = tid; e < 48 * 32; e += 128) {   // V^T tile: [d][n_local]
      int d = e >> 5, nl = e & 31;
      int n = ks * 32 + nl;
      _Float16 v = (_Float16)0.0f;
      if (n < NTOK)
        v = kvf[((size_t)b * NPOS + idx_sub[n]) * (2 * CDIM) + CDIM + h * HDIM + d];
      lVt[d * 40 + nl] = v;
    }
    __syncthreads();
    if (wave < 3) {
      v16h a  = frag_a16(lP + ks * 32, 224, lane);
      v16h bb = frag_b16(lVt + wave * 16 * 40, 40, lane);
      o = wmma_f16(a, bb, o);
    }
    __syncthreads();
  }
  if (wave < 3) {
    const int d  = wave * 16 + (lane & 15);
    const int mr = (lane & 16) ? 8 : 0;
#pragma unroll
    for (int r = 0; r < 8; ++r) {
      int m = m0 + r + mr;
      if (m < NTOK)
        qc0[((size_t)b * NTOK + m) * CDIM + h * HDIM + d] = (_Float16)o[r];
    }
  }
}

// ---------------------------------------------------------------------------
// attn2 + argmax + window-mask bitmap.  grid (13 m-tiles, B), block 128.
// Head-mean accumulated in LDS (deterministic); argmax -> centroid -> bitmap.
// ---------------------------------------------------------------------------
__global__ __launch_bounds__(128)
void attn2_mask_kernel(const _Float16* __restrict__ qc,   // (B*196)x384
                       const _Float16* __restrict__ kvf,
                       const int* __restrict__ idxs,
                       unsigned* __restrict__ maskbits) { // (B*196)x98
  const int m0 = blockIdx.x * 16;
  const int b  = blockIdx.y;
  const int tid = threadIdx.x, lane = tid & 31, wave = tid >> 5;

  __shared__ _Float16 lQ[16 * 64];
  __shared__ _Float16 lK[4][16 * 64];
  __shared__ float    lS[16 * 224];
  __shared__ int      cent[16];
  __shared__ unsigned sbits[MASK_WORDS];

  for (int e = tid; e < 16 * 224; e += 128) lS[e] = 0.0f;
  __syncthreads();

  for (int h = 0; h < HEADS; ++h) {
    load_q_tile(lQ, qc + (size_t)b * NTOK * CDIM + h * HDIM, m0, tid);
    __syncthreads();
    for (int g = 0; g < 4; ++g) {
      int tn = g * 4 + wave;
      if (tn < 13) {
        load_k_tile_gather(lK[wave], kvf, idxs, b, h, tn * 16, lane);
        v8f s = score_tile(lQ, lK[wave], lane);
        const int ncol = tn * 16 + (lane & 15);
        const int mr   = (lane & 16) ? 8 : 0;
#pragma unroll
        for (int r = 0; r < 8; ++r)
          lS[(r + mr) * 224 + ncol] += s[r] * (1.0f / HEADS);
      }
      __syncthreads();
    }
  }

  // argmax over n (first max wins, like jnp.argmax)
  if (tid < 16) {
    float best = lS[tid * 224];
    int bi = 0;
    for (int n = 1; n < NTOK; ++n) {
      float v = lS[tid * 224 + n];
      if (v > best) { best = v; bi = n; }
    }
    cent[tid] = idxs[bi];
  }
  __syncthreads();

  // 13x13 window bitmap: offsets = col - 56*row, row,col in [-6,6], clipped
  for (int r = 0; r < 16; ++r) {
    for (int e = tid; e < MASK_WORDS; e += 128) sbits[e] = 0u;
    __syncthreads();
    if (tid < 169) {
      int row = tid / 13 - 6;
      int col = tid % 13 - 6;
      int adj = cent[r] + col - 56 * row;
      adj = adj < 0 ? 0 : (adj > NPOS - 1 ? NPOS - 1 : adj);
      atomicOr(&sbits[adj >> 5], 1u << (adj & 31));
    }
    __syncthreads();
    int m = m0 + r;
    if (m < NTOK)
      for (int e = tid; e < MASK_WORDS; e += 128)
        maskbits[((size_t)b * NTOK + m) * MASK_WORDS + e] = sbits[e];
    __syncthreads();
  }
}

// ---------------------------------------------------------------------------
// attn3: flash attention over full N=3136 with bitmap mask; out f16.
// grid (13 m-tiles, H, B), block 128 (4 waves); 32-column chunks.
// K chunk staged by the Tensor Data Mover (zero padding via TDM OOB rule).
// ---------------------------------------------------------------------------
__global__ __launch_bounds__(128)
void attn3_kernel(const _Float16* __restrict__ qc,
                  const _Float16* __restrict__ kvf,
                  const unsigned* __restrict__ maskbits,
                  _Float16* __restrict__ aout) {
  const int m0 = blockIdx.x * 16;
  const int h  = blockIdx.y;
  const int b  = blockIdx.z;
  const int tid = threadIdx.x, lane = tid & 31, wave = tid >> 5;

  __shared__ _Float16 lQ[16 * 64];
  __shared__ _Float16 lK[32 * 64];
  __shared__ _Float16 lVt[48 * 40];
  __shared__ float    lS[16 * 32];
  __shared__ _Float16 lP[16 * 32];
  __shared__ float    rowM[16], rowL[16], rowC[16];

  load_q_tile(lQ, qc + (size_t)b * NTOK * CDIM + h * HDIM, m0, tid);
  if (tid < 16) { rowM[tid] = -1e30f; rowL[tid] = 0.0f; }
  __syncthreads();

  v8f o = vzero8();
  for (int n0 = 0; n0 < NPOS; n0 += 32) {
    for (int e = tid; e < 48 * 32; e += 128) {          // V^T chunk (transpose)
      int d = e >> 5, nl = e & 31;
      lVt[d * 40 + nl] =
          kvf[((size_t)b * NPOS + n0 + nl) * (2 * CDIM) + CDIM + h * HDIM + d];
    }
#if HAVE_TDM
    if (wave == 0) {                                    // 32x48 tile, pad to 64
      unsigned la = (unsigned)(size_t)(void*)lK;
      unsigned long long ga = (unsigned long long)(size_t)(
          kvf + ((size_t)b * NPOS + n0) * (2 * CDIM) + h * HDIM);
      tdm_load_2d_f16(la, ga, /*tile_w=*/64, /*tile_h=*/32,
                      /*tensor_w=*/HDIM, /*row_stride=*/2 * CDIM);
      __builtin_amdgcn_s_wait_tensorcnt((short)0);
    }
#else
    for (int e = tid; e < 32 * 8; e += 128) {           // K chunk, b128 copies
      const uint4 z = {0u, 0u, 0u, 0u};
      int row = e >> 3, c8 = (e & 7) * 8;
      uint4 v = z;
      if (c8 < HDIM)
        v = *(const uint4*)(kvf + ((size_t)b * NPOS + n0 + row) * (2 * CDIM) +
                            h * HDIM + c8);
      *(uint4*)(lK + row * 64 + c8) = v;
    }
#endif
    __syncthreads();

    if (wave < 2) {                                     // scores + mask
      v8f s = score_tile(lQ, lK + wave * 16 * 64, lane);
      const int nl = wave * 16 + (lane & 15);
      const int n  = n0 + nl;
      const int mr = (lane & 16) ? 8 : 0;
#pragma unroll
      for (int r = 0; r < 8; ++r) {
        int m = m0 + r + mr;
        int mc = m < NTOK ? m : NTOK - 1;
        unsigned w = maskbits[((size_t)b * NTOK + mc) * MASK_WORDS + (n >> 5)];
        float msk = ((w >> (n & 31)) & 1u) ? 0.0f : NEG_BIG;
        lS[(r + mr) * 32 + nl] = s[r] * QK_SCALE + msk;
      }
    }
    __syncthreads();

    if (tid < 16) {                                     // online softmax update
      float mold = rowM[tid], mx = mold;
      for (int c = 0; c < 32; ++c) mx = fmaxf(mx, lS[tid * 32 + c]);
      float corr = __expf(mold - mx);
      float sum = 0.0f;
      for (int c = 0; c < 32; ++c) {
        float p = __expf(lS[tid * 32 + c] - mx);
        lP[tid * 32 + c] = (_Float16)p;
        sum += p;
      }
      rowL[tid] = rowL[tid] * corr + sum;
      rowM[tid] = mx;
      rowC[tid] = corr;
    }
    __syncthreads();

    if (wave < 3) {                                     // O = O*corr + P@V
      const int mr = (lane & 16) ? 8 : 0;
#pragma unroll
      for (int r = 0; r < 8; ++r) o[r] *= rowC[r + mr];
      v16h a  = frag_a16(lP, 32, lane);
      v16h bb = frag_b16(lVt + wave * 16 * 40, 40, lane);
      o = wmma_f16(a, bb, o);
    }
    __syncthreads();
  }

  if (wave < 3) {
    const int d  = wave * 16 + (lane & 15);
    const int mr = (lane & 16) ? 8 : 0;
#pragma unroll
    for (int r = 0; r < 8; ++r) {
      int m = m0 + r + mr;
      if (m < NTOK)
        aout[((size_t)b * NTOK + m) * CDIM + h * HDIM + d] =
            (_Float16)(o[r] / rowL[r + mr]);
    }
  }
}

// ---------------------------------------------------------------------------
// Launcher
// ---------------------------------------------------------------------------
static inline size_t align256(size_t x) { return (x + 255) & ~(size_t)255; }

extern "C" void kernel_launch(void* const* d_in, const int* in_sizes, int n_in,
                              void* d_out, int out_size, void* d_ws, size_t ws_size,
                              hipStream_t stream) {
  (void)in_sizes; (void)n_in; (void)out_size; (void)ws_size;
  const float* x      = (const float*)d_in[0];
  const float* q      = (const float*)d_in[1];
  const float* kv_w   = (const float*)d_in[2];
  const float* proj_w = (const float*)d_in[3];
  const float* proj_b = (const float*)d_in[4];
  const float* ln_w   = (const float*)d_in[5];
  const float* ln_b   = (const float*)d_in[6];
  const float* fc1_w  = (const float*)d_in[7];
  const float* fc1_b  = (const float*)d_in[8];
  const float* fc2_w  = (const float*)d_in[9];
  const float* fc2_b  = (const float*)d_in[10];
  const int* idx_sub  = (const int*)d_in[11];
  const int* idxs     = (const int*)d_in[12];
  float* out = (float*)d_out;

  const int M1 = BATCH * NPOS;   // 100352
  const int M2 = BATCH * NTOK;   // 6272

  char* ws = (char*)d_ws;
  size_t off = 0;
  auto take = [&](size_t bytes) { char* p = ws + off; off = align256(off + bytes); return p; };
  _Float16* xn16    = (_Float16*)take((size_t)M1 * CDIM * 2);
  _Float16* kvw16   = (_Float16*)take((size_t)2 * CDIM * CDIM * 2);
  _Float16* kv16    = (_Float16*)take((size_t)M1 * 2 * CDIM * 2);
  _Float16* qs16    = (_Float16*)take((size_t)NTOK * CDIM * 2);
  _Float16* qc016   = (_Float16*)take((size_t)M2 * CDIM * 2);
  _Float16* fc1w16  = (_Float16*)take((size_t)2 * CDIM * CDIM * 2);
  _Float16* fc2w16  = (_Float16*)take((size_t)CDIM * 2 * CDIM * 2);
  _Float16* projw16 = (_Float16*)take((size_t)CDIM * CDIM * 2);
  _Float16* h16     = (_Float16*)take((size_t)M2 * 2 * CDIM * 2);
  _Float16* qcond16 = (_Float16*)take((size_t)M2 * CDIM * 2);
  unsigned* mbits   = (unsigned*)take((size_t)M2 * MASK_WORDS * 4);
  _Float16* aout16  = (_Float16*)take((size_t)M2 * CDIM * 2);

  int nkvw = 2 * CDIM * CDIM;
  cvt_f32_to_f16<<<(nkvw + 255) / 256, 256, 0, stream>>>(kv_w, kvw16, nkvw, 1.0f);
  cvt_f32_to_f16<<<(nkvw + 255) / 256, 256, 0, stream>>>(fc1_w, fc1w16, nkvw, 1.0f);
  cvt_f32_to_f16<<<(nkvw + 255) / 256, 256, 0, stream>>>(fc2_w, fc2w16, nkvw, 1.0f);
  int nproj = CDIM * CDIM;
  cvt_f32_to_f16<<<(nproj + 255) / 256, 256, 0, stream>>>(proj_w, projw16, nproj, 1.0f);
  int nq = NTOK * CDIM;
  cvt_f32_to_f16<<<(nq + 255) / 256, 256, 0, stream>>>(q, qs16, nq, QK_SCALE);

  ln_kernel<<<M1, 128, 0, stream>>>(x, ln_w, ln_b, xn16);

  gemm_nt_f16<<<dim3(2 * CDIM / 64, M1 / 128), 256, 0, stream>>>(
      xn16, kvw16, nullptr, kv16, nullptr, M1, CDIM, 2 * CDIM, 0);

  attn1_kernel<<<dim3(13, HEADS, BATCH), 128, 0, stream>>>(qs16, kv16, idx_sub, qc016);

  gemm_nt_f16<<<dim3(2 * CDIM / 64, M2 / 128), 256, 0, stream>>>(
      qc016, fc1w16, fc1_b, h16, nullptr, M2, CDIM, 2 * CDIM, 1);
  gemm_nt_f16<<<dim3(CDIM / 64, M2 / 128), 256, 0, stream>>>(
      h16, fc2w16, fc2_b, qcond16, nullptr, M2, 2 * CDIM, CDIM, 0);

  attn2_mask_kernel<<<dim3(13, BATCH), 128, 0, stream>>>(qcond16, kv16, idxs, mbits);

  attn3_kernel<<<dim3(13, HEADS, BATCH), 128, 0, stream>>>(qcond16, kv16, mbits, aout16);

  gemm_nt_f16<<<dim3(CDIM / 64, M2 / 128), 256, 0, stream>>>(
      aout16, projw16, proj_b, nullptr, out, M2, CDIM, CDIM, 0);
}
